// SimpleDenoisingNetwork_33921651704335
// MI455X (gfx1250) — matrix operand order repaired and checked by reading, not compiled, not run on hardware
//
#include <hip/hip_runtime.h>
#include <hip/hip_bf16.h>
#include <math.h>

#define C_S    384
#define N_TOK  1024
#define N_ATOM 24576
#define FDIM   256
#define NH     4
#define HD     96
#define QKVN   (3 * C_S)
#define FFH    (4 * C_S)

typedef __bf16 v16bf __attribute__((ext_vector_type(16)));
typedef float  v8f   __attribute__((ext_vector_type(8)));
typedef unsigned int v8u __attribute__((ext_vector_type(8)));

union FragU { v8u u; v16bf h; };

__device__ __forceinline__ unsigned short f2bf(float f) {
  union { float f; unsigned u; } cv; cv.f = f;
  unsigned u = cv.u;
  u += 0x7FFFu + ((u >> 16) & 1u);   // round-to-nearest-even
  return (unsigned short)(u >> 16);
}

__device__ __forceinline__ void gemm_epilogue(
    v8f acc, int m0, int n, bool valid,
    const float* __restrict__ bias, float* __restrict__ Cf,
    unsigned short* __restrict__ Cb, const float* __restrict__ Res,
    int ldc, long long zoff, float alpha, int act)
{
  if (!valid) return;
  const float bv = bias ? bias[n] : 0.0f;
#pragma unroll
  for (int r = 0; r < 8; ++r) {
    float v = acc[r] * alpha + bv;
    if (act == 1) v = v / (1.0f + __expf(-v));   // SiLU
    long long off = zoff + (long long)(m0 + r) * ldc + n;
    if (Res) v += Res[off];
    if (Cf) Cf[off] = v;
    if (Cb) Cb[off] = f2bf(v);
  }
}

// ---------------------------------------------------------------------------
// bf16 WMMA GEMM:  C[z][m][n] = act(alpha * sum_k A[z][m][k]*W[z][n][k] + bias[n]) (+Res)
// A: [M,K] bf16 row-major (lda, +z*aZ), W: [N,K] bf16 row-major (ldw, +z*wZ).
// Block = 128 threads = 4 waves in a 2x2 grid; each wave owns a 32x32 C tile
// (2x2 WMMA fragments, 4 accumulators) -> 64x64 block tile, 16 FLOP/byte of
// issued vmem traffic. K-chunk-ahead global_prefetch_b8 for latency hiding.
// Requires: M % 64 == 0, N % 16 == 0, K % 32 == 0 (N edge handled by clamping
// loads in-bounds and skipping stores, keeping EXEC all-ones for WMMA).
// ---------------------------------------------------------------------------
__global__ void gemm_bf16_wmma(
    const unsigned short* __restrict__ A, int lda, long long aZ,
    const unsigned short* __restrict__ W, int ldw, long long wZ,
    const float* __restrict__ bias,
    float* __restrict__ Cf, unsigned short* __restrict__ Cb,
    int ldc, long long cZ,
    const float* __restrict__ Res,
    float alpha, int K, int N, int act)
{
  const int z    = blockIdx.z;
  const int lane = threadIdx.x & 31;
  const int wid  = threadIdx.x >> 5;          // 0..3
  const int half = (lane >> 4) & 1;
  const int l    = lane & 15;
  const int tileM  = blockIdx.y * 64 + (wid >> 1) * 32;
  const int tileN  = blockIdx.x * 64 + (wid & 1) * 32;
  const int tn0 = tileN, tn1 = tileN + 16;
  const bool valid0 = tn0 < N, valid1 = tn1 < N;

  // A fragment rows (two M tiles)
  const unsigned short* a0 = A + (long long)z * aZ + (long long)(tileM + l) * lda;
  const unsigned short* a1 = a0 + 16 * lda;
  // W fragment rows (two N tiles); clamp so out-of-range tiles still load in-bounds
  const int cn0 = valid0 ? (tn0 + l) : (N - 1);
  const int cn1 = valid1 ? (tn1 + l) : (N - 1);
  const unsigned short* w0 = W + (long long)z * wZ + (long long)cn0 * ldw;
  const unsigned short* w1 = W + (long long)z * wZ + (long long)cn1 * ldw;

  v8f acc00 = {}, acc01 = {}, acc10 = {}, acc11 = {};
  for (int k0 = 0; k0 < K; k0 += 32) {
    if (k0 + 128 < K) {                        // gfx1250 global_prefetch_b8
      __builtin_prefetch(a0 + k0 + 128, 0, 1);
      __builtin_prefetch(a1 + k0 + 128, 0, 1);
      __builtin_prefetch(w0 + k0 + 128, 0, 1);
      __builtin_prefetch(w1 + k0 + 128, 0, 1);
    }
    FragU fa0, fa1, fb0, fb1;
#pragma unroll
    for (int p = 0; p < 8; ++p) {
      // A fragment (16x32): lanes 0-15 K pairs {0..7,16..23}; lanes 16-31 {8..15,24..31}
      int ka = k0 + ((p < 4) ? 0 : 16) + half * 8 + 2 * (p & 3);
      fa0.u[p] = *(const unsigned int*)(a0 + ka);
      fa1.u[p] = *(const unsigned int*)(a1 + ka);
      // B fragment (32x16): lanes 0-15 K pairs {0..7,16..23}; lanes 16-31 {8..15,24..31}
      int kb = k0 + ((p < 4) ? 0 : 16) + half * 8 + 2 * (p & 3);
      fb0.u[p] = *(const unsigned int*)(w0 + kb);
      fb1.u[p] = *(const unsigned int*)(w1 + kb);
    }
    acc00 = __builtin_amdgcn_wmma_f32_16x16x32_bf16(false, fa0.h, false, fb0.h, (short)0, acc00, false, false);
    acc01 = __builtin_amdgcn_wmma_f32_16x16x32_bf16(false, fa0.h, false, fb1.h, (short)0, acc01, false, false);
    acc10 = __builtin_amdgcn_wmma_f32_16x16x32_bf16(false, fa1.h, false, fb0.h, (short)0, acc10, false, false);
    acc11 = __builtin_amdgcn_wmma_f32_16x16x32_bf16(false, fa1.h, false, fb1.h, (short)0, acc11, false, false);
  }

  const long long zoff = (long long)z * cZ;
  const int m0 = tileM + half * 8;             // C/D layout: VGPR r = row r, +8 for lanes 16-31
  gemm_epilogue(acc00, m0,      tn0 + l, valid0, bias, Cf, Cb, Res, ldc, zoff, alpha, act);
  gemm_epilogue(acc01, m0,      tn1 + l, valid1, bias, Cf, Cb, Res, ldc, zoff, alpha, act);
  gemm_epilogue(acc10, m0 + 16, tn0 + l, valid0, bias, Cf, Cb, Res, ldc, zoff, alpha, act);
  gemm_epilogue(acc11, m0 + 16, tn1 + l, valid1, bias, Cf, Cb, Res, ldc, zoff, alpha, act);
}

// ---------------------------------------------------------------------------
// Diffusion conditioning: fourier embed (256) -> noise (256x256 GEMV) -> ns proj (384x256)
// Single block, everything in LDS. z-branch of the reference is dead code -> skipped.
// ---------------------------------------------------------------------------
__global__ void cond_kernel(const float* __restrict__ sigma,
                            const float* __restrict__ fw, const float* __restrict__ fb,
                            const float* __restrict__ nsw, const float* __restrict__ nsb,
                            float* __restrict__ ns_proj)
{
  __shared__ float emb[FDIM];
  __shared__ float noise[FDIM];
  const int tid = threadIdx.x;
  if (tid < FDIM / 2) {
    float f  = __expf(-logf(1000.0f) * (float)tid / (float)(FDIM / 2));
    float xf = sigma[0] * f;
    emb[tid]            = cosf(xf);
    emb[tid + FDIM / 2] = sinf(xf);
  }
  __syncthreads();
  {
    float accv = fb[tid];
    const float* wr = fw + (long long)tid * FDIM;
    for (int i = 0; i < FDIM; ++i) accv += emb[i] * wr[i];
    noise[tid] = accv;
  }
  __syncthreads();
  for (int j = tid; j < C_S; j += 256) {
    float accv = nsb[j];
    const float* wr = nsw + (long long)j * FDIM;
    for (int i = 0; i < FDIM; ++i) accv += noise[i] * wr[i];
    ns_proj[j] = accv;
  }
}

// LayerNorm over C columns, optional broadcast-add vector, f32 and/or bf16 output.
__global__ void ln_kernel(const float* __restrict__ X, int C,
                          const float* __restrict__ g, const float* __restrict__ be,
                          const float* __restrict__ addv,
                          float* __restrict__ outF, unsigned short* __restrict__ outB)
{
  __shared__ float sbuf[128];
  const int row = blockIdx.x;
  const int tid = threadIdx.x;
  const float* x = X + (long long)row * C;
  float s = 0.f;
  for (int c = tid; c < C; c += 128) s += x[c];
  sbuf[tid] = s; __syncthreads();
  for (int k = 64; k > 0; k >>= 1) { if (tid < k) sbuf[tid] += sbuf[tid + k]; __syncthreads(); }
  const float mean = sbuf[0] / (float)C;
  __syncthreads();
  float v = 0.f;
  for (int c = tid; c < C; c += 128) { float d = x[c] - mean; v += d * d; }
  sbuf[tid] = v; __syncthreads();
  for (int k = 64; k > 0; k >>= 1) { if (tid < k) sbuf[tid] += sbuf[tid + k]; __syncthreads(); }
  const float rstd = rsqrtf(sbuf[0] / (float)C + 1e-5f);
  for (int c = tid; c < C; c += 128) {
    float y = (x[c] - mean) * rstd * g[c] + be[c];
    if (addv) y += addv[c];
    long long off = (long long)row * C + c;
    if (outF) outF[off] = y;
    if (outB) outB[off] = f2bf(y);
  }
}

__global__ void zero_kernel(float* __restrict__ p, int n)
{
  int i = blockIdx.x * blockDim.x + threadIdx.x;
  if (i < n) p[i] = 0.f;
}

__global__ void cvt_bf16_kernel(const float* __restrict__ src, unsigned short* __restrict__ dst, int n)
{
  int i = blockIdx.x * blockDim.x + threadIdx.x;
  if (i < n) dst[i] = f2bf(src[i]);
}

// coord_feat = x @ ce_w.T + ce_b (K=3, VALU), scattered with f32 atomics into per-token sums
__global__ void atom_scatter_kernel(const float* __restrict__ x, const int* __restrict__ idx,
                                    const float* __restrict__ cw, const float* __restrict__ cb,
                                    float* __restrict__ seg, float* __restrict__ cnt)
{
  long long gid = (long long)blockIdx.x * blockDim.x + threadIdx.x;
  if (gid >= (long long)N_ATOM * C_S) return;
  int a = (int)(gid / C_S), c = (int)(gid % C_S);
  int t = idx[a];
  float x0 = x[a * 3], x1 = x[a * 3 + 1], x2 = x[a * 3 + 2];
  float v = cb[c] + x0 * cw[c * 3] + x1 * cw[c * 3 + 1] + x2 * cw[c * 3 + 2];
  atomicAdd(&seg[(long long)t * C_S + c], v);
  if (c == 0) atomicAdd(&cnt[t], 1.0f);
}

__global__ void token_combine_kernel(const float* __restrict__ seg, const float* __restrict__ cnt,
                                     const float* __restrict__ scond,
                                     float* __restrict__ tf, unsigned short* __restrict__ tb)
{
  int gid = blockIdx.x * blockDim.x + threadIdx.x;
  if (gid >= N_TOK * C_S) return;
  int t = gid / C_S;
  float v = seg[gid] / fmaxf(cnt[t], 1.0f) + scond[gid];
  tf[gid] = v;
  tb[gid] = f2bf(v);
}

// vt[h][d][t] = V[t][h*HD+d]  (so attn@V becomes the W[N,K]-row-major GEMM pattern)
__global__ void v_repack_kernel(const unsigned short* __restrict__ qkv, unsigned short* __restrict__ vt)
{
  int gid = blockIdx.x * blockDim.x + threadIdx.x;
  if (gid >= NH * HD * N_TOK) return;
  int h = gid / (HD * N_TOK);
  int r = gid % (HD * N_TOK);
  int d = r / N_TOK, t = r % N_TOK;
  vt[gid] = qkv[(long long)t * QKVN + 2 * C_S + h * HD + d];
}

__global__ void softmax_kernel(const float* __restrict__ S, unsigned short* __restrict__ P, int n)
{
  __shared__ float sbuf[256];
  const long long row = blockIdx.x;
  const float* x = S + row * n;
  unsigned short* p = P + row * n;
  const int tid = threadIdx.x;
  float m = -3.4e38f;
  for (int c = tid; c < n; c += 256) m = fmaxf(m, x[c]);
  sbuf[tid] = m; __syncthreads();
  for (int k = 128; k > 0; k >>= 1) { if (tid < k) sbuf[tid] = fmaxf(sbuf[tid], sbuf[tid + k]); __syncthreads(); }
  m = sbuf[0]; __syncthreads();
  float s = 0.f;
  for (int c = tid; c < n; c += 256) s += __expf(x[c] - m);
  sbuf[tid] = s; __syncthreads();
  for (int k = 128; k > 0; k >>= 1) { if (tid < k) sbuf[tid] += sbuf[tid + k]; __syncthreads(); }
  const float inv = 1.0f / sbuf[0];
  for (int c = tid; c < n; c += 256) p[c] = f2bf(__expf(x[c] - m) * inv);
}

// token -> atom gather + coord head (N=3, K=384; VALU; token rows live in L2)
__global__ void coord_out_kernel(const float* __restrict__ tok, const int* __restrict__ idx,
                                 const float* __restrict__ cw, const float* __restrict__ cb,
                                 float* __restrict__ out)
{
  int gid = blockIdx.x * blockDim.x + threadIdx.x;
  if (gid >= N_ATOM * 3) return;
  int a = gid / 3, c = gid % 3;
  const float* row = tok + (long long)idx[a] * C_S;
  const float* w   = cw + c * C_S;
  float accv = cb[c];
  for (int k = 0; k < C_S; ++k) accv += row[k] * w[k];
  out[gid] = accv;
}

// ---------------------------------------------------------------------------
extern "C" void kernel_launch(void* const* d_in, const int* in_sizes, int n_in,
                              void* d_out, int out_size, void* d_ws, size_t ws_size,
                              hipStream_t stream) {
  (void)in_sizes; (void)n_in; (void)out_size;
  const float* x_noisy  = (const float*)d_in[0];
  const float* s_in     = (const float*)d_in[1];
  /* d_in[2] = z: 537 MB, dead code in the reference -> never touched */
  const float* sigma    = (const float*)d_in[3];
  const int*   a2t      = (const int*)d_in[4];
  const float* fourier_w = (const float*)d_in[5];
  const float* fourier_b = (const float*)d_in[6];
  const float* ns_w = (const float*)d_in[7];
  const float* ns_b = (const float*)d_in[8];
  const float* lns_g = (const float*)d_in[11];
  const float* lns_b = (const float*)d_in[12];
  const float* ce_w = (const float*)d_in[15];
  const float* ce_b = (const float*)d_in[16];
  const float* in_w = (const float*)d_in[17];
  const float* in_b = (const float*)d_in[18];
  const float* out_w = (const float*)d_in[19];
  const float* out_b = (const float*)d_in[20];
  const float* ffln_g = (const float*)d_in[21];
  const float* ffln_b = (const float*)d_in[22];
  const float* ff1_w = (const float*)d_in[23];
  const float* ff1_b = (const float*)d_in[24];
  const float* ff2_w = (const float*)d_in[25];
  const float* ff2_b = (const float*)d_in[26];
  const float* co_w = (const float*)d_in[27];
  const float* co_b = (const float*)d_in[28];
  float* out = (float*)d_out;

  // ---- workspace carve (256B aligned) ----
  size_t off = 0;
  char* base = (char*)d_ws;
  auto carve = [&](size_t bytes) -> char* {
    off = (off + 255) & ~(size_t)255;
    char* p = base + off;
    off += bytes;
    return p;
  };
  float* ns_proj   = (float*)carve(C_S * 4);
  float* seg_sum   = (float*)carve((size_t)N_TOK * C_S * 4 + N_TOK * 4); // counts adjacent
  float* counts    = seg_sum + (size_t)N_TOK * C_S;
  float* s_cond    = (float*)carve((size_t)N_TOK * C_S * 4);
  float* token_f   = (float*)carve((size_t)N_TOK * C_S * 4);
  unsigned short* token_b = (unsigned short*)carve((size_t)N_TOK * C_S * 2);
  unsigned short* in_w_b  = (unsigned short*)carve((size_t)QKVN * C_S * 2);
  unsigned short* out_w_b = (unsigned short*)carve((size_t)C_S * C_S * 2);
  unsigned short* ff1_w_b = (unsigned short*)carve((size_t)FFH * C_S * 2);
  unsigned short* ff2_w_b = (unsigned short*)carve((size_t)C_S * FFH * 2);
  unsigned short* qkv_b   = (unsigned short*)carve((size_t)N_TOK * QKVN * 2);
  unsigned short* vt_b    = (unsigned short*)carve((size_t)NH * HD * N_TOK * 2);
  float* scores           = (float*)carve((size_t)NH * N_TOK * N_TOK * 4);
  unsigned short* attn_b  = (unsigned short*)carve((size_t)NH * N_TOK * N_TOK * 2);
  unsigned short* ao_b    = (unsigned short*)carve((size_t)N_TOK * C_S * 2);
  float* token2_f         = (float*)carve((size_t)N_TOK * C_S * 4);
  unsigned short* hln_b   = (unsigned short*)carve((size_t)N_TOK * C_S * 2);
  unsigned short* ffh_b   = (unsigned short*)carve((size_t)N_TOK * FFH * 2);
  float* token3_f         = (float*)carve((size_t)N_TOK * C_S * 4);
  if (off > ws_size) return;  // scratch too small: deterministic no-op

  // 1) zero segment sums + counts
  {
    int n = N_TOK * C_S + N_TOK;
    zero_kernel<<<(n + 255) / 256, 256, 0, stream>>>(seg_sum, n);
  }
  // 2) conditioning -> ns_proj[384]
  cond_kernel<<<1, 256, 0, stream>>>(sigma, fourier_w, fourier_b, ns_w, ns_b, ns_proj);
  // 3) s_cond = LN(s)*g+b + ns_proj
  ln_kernel<<<N_TOK, 128, 0, stream>>>(s_in, C_S, lns_g, lns_b, ns_proj, s_cond, nullptr);
  // 4) weights -> bf16
  { int n = QKVN * C_S; cvt_bf16_kernel<<<(n + 255) / 256, 256, 0, stream>>>(in_w,  in_w_b,  n); }
  { int n = C_S * C_S;  cvt_bf16_kernel<<<(n + 255) / 256, 256, 0, stream>>>(out_w, out_w_b, n); }
  { int n = FFH * C_S;  cvt_bf16_kernel<<<(n + 255) / 256, 256, 0, stream>>>(ff1_w, ff1_w_b, n); }
  { int n = C_S * FFH;  cvt_bf16_kernel<<<(n + 255) / 256, 256, 0, stream>>>(ff2_w, ff2_w_b, n); }
  // 5) atom -> token segment sums (+counts)
  {
    long long n = (long long)N_ATOM * C_S;
    atom_scatter_kernel<<<(unsigned)((n + 255) / 256), 256, 0, stream>>>(
        x_noisy, a2t, ce_w, ce_b, seg_sum, counts);
  }
  // 6) token_feat = seg_mean + s_cond (f32 + bf16)
  {
    int n = N_TOK * C_S;
    token_combine_kernel<<<(n + 255) / 256, 256, 0, stream>>>(seg_sum, counts, s_cond, token_f, token_b);
  }
  // 7) qkv = token_feat @ in_w.T + in_b  -> bf16 [1024,1152]
  gemm_bf16_wmma<<<dim3(QKVN / 64, N_TOK / 64, 1), 128, 0, stream>>>(
      token_b, C_S, 0, in_w_b, C_S, 0, in_b,
      nullptr, qkv_b, QKVN, 0, nullptr, 1.0f, C_S, QKVN, 0);
  // 8) V transpose repack
  {
    int n = NH * HD * N_TOK;
    v_repack_kernel<<<(n + 255) / 256, 256, 0, stream>>>(qkv_b, vt_b);
  }
  // 9) scores[h] = Q_h @ K_h.T / sqrt(96)   (z = head)
  gemm_bf16_wmma<<<dim3(N_TOK / 64, N_TOK / 64, NH), 128, 0, stream>>>(
      qkv_b, QKVN, HD, qkv_b + C_S, QKVN, HD, nullptr,
      scores, nullptr, N_TOK, (long long)N_TOK * N_TOK, nullptr,
      0.1020620726159658f /* 1/sqrt(96) */, HD, N_TOK, 0);
  // 10) softmax rows -> attn bf16
  softmax_kernel<<<NH * N_TOK, 256, 0, stream>>>(scores, attn_b, N_TOK);
  // 11) ao[:, h*96:(h+1)*96] = attn_h @ V_h   (N=96 -> edge-clamped tiles)
  gemm_bf16_wmma<<<dim3((HD + 63) / 64, N_TOK / 64, NH), 128, 0, stream>>>(
      attn_b, N_TOK, (long long)N_TOK * N_TOK, vt_b, N_TOK, (long long)HD * N_TOK, nullptr,
      nullptr, ao_b, C_S, HD, nullptr, 1.0f, N_TOK, HD, 0);
  // 12) token2 = token_feat + ao @ out_w.T + out_b
  gemm_bf16_wmma<<<dim3(C_S / 64, N_TOK / 64, 1), 128, 0, stream>>>(
      ao_b, C_S, 0, out_w_b, C_S, 0, out_b,
      token2_f, nullptr, C_S, 0, token_f, 1.0f, C_S, C_S, 0);
  // 13) FFN layernorm -> bf16
  ln_kernel<<<N_TOK, 128, 0, stream>>>(token2_f, C_S, ffln_g, ffln_b, nullptr, nullptr, hln_b);
  // 14) ffh = silu(hln @ ff1_w.T + ff1_b) -> bf16 [1024,1536]
  gemm_bf16_wmma<<<dim3(FFH / 64, N_TOK / 64, 1), 128, 0, stream>>>(
      hln_b, C_S, 0, ff1_w_b, C_S, 0, ff1_b,
      nullptr, ffh_b, FFH, 0, nullptr, 1.0f, C_S, FFH, 1);
  // 15) token3 = token2 + ffh @ ff2_w.T + ff2_b
  gemm_bf16_wmma<<<dim3(C_S / 64, N_TOK / 64, 1), 128, 0, stream>>>(
      ffh_b, FFH, 0, ff2_w_b, FFH, 0, ff2_b,
      token3_f, nullptr, C_S, 0, token2_f, 1.0f, FFH, C_S, 0);
  // 16) gather + coord head -> d_out [24576,3]
  {
    int n = N_ATOM * 3;
    coord_out_kernel<<<(n + 255) / 256, 256, 0, stream>>>(token3_f, a2t, co_w, co_b, out);
  }
}
